// AIFI_SLA_61435212202334
// MI455X (gfx1250) — matrix operand order repaired and checked by reading, not compiled, not run on hardware
//
#include <hip/hip_runtime.h>
#include <hip/hip_bf16.h>

typedef unsigned int u32;
typedef unsigned long long u64;
typedef __attribute__((ext_vector_type(16))) __bf16 v16bf;
typedef __attribute__((ext_vector_type(8)))  __bf16 v8bf;
typedef __attribute__((ext_vector_type(8)))  float  v8f;
typedef __attribute__((ext_vector_type(4)))  float  v4f;

#define EPS 1e-5f

// ---------------------------------------------------------------------------
// WMMA bf16 16x16x32, f32 accum. Fragment layouts per CDNA5 ISA 7.12.2.
// Every fragment = 2x ds_load_b128 (LDS strides are multiples of 8 bf16,
// bases 16B aligned).
// ---------------------------------------------------------------------------
__device__ __forceinline__ v8f wmma_bf16(v16bf a, v16bf b, v8f c) {
  return __builtin_amdgcn_wmma_f32_16x16x32_bf16(false, a, false, b,
                                                 (short)0, c, false, false);
}

__device__ __forceinline__ v16bf frag_from2(const __bf16* p0, const __bf16* p1) {
  v8bf a = *(const v8bf*)p0;
  v8bf b = *(const v8bf*)p1;
  return __builtin_shufflevector(a, b, 0, 1, 2, 3, 4, 5, 6, 7,
                                 8, 9, 10, 11, 12, 13, 14, 15);
}

__device__ __forceinline__ v16bf frag_a(const __bf16* tile, int stride, int lane) {
  const __bf16* p = tile + (lane & 15) * stride + ((lane >> 4) << 3);
  return frag_from2(p, p + 16);
}

__device__ __forceinline__ v16bf frag_b(const __bf16* tile, int stride, int lane) {
  const __bf16* p = tile + (lane & 15) * stride + ((lane >> 4) << 3);
  return frag_from2(p, p + 16);
}

// ---------------------------------------------------------------------------
// Tensor Data Mover: 2D tile (global -> LDS) with optional LDS row padding.
// D# built per CDNA5 ISA 8.3/8.4 (group0: count/lds/global/type; group1:
// data_size=2B, pad fields, dims/strides). Fallback: manual uint4 staging.
// ---------------------------------------------------------------------------
#if defined(__has_builtin)
#if __has_builtin(__builtin_amdgcn_tensor_load_to_lds) && \
    __has_builtin(__builtin_amdgcn_s_wait_tensorcnt)
#define HAVE_TDM 1
#endif
#endif

#ifdef HAVE_TDM
typedef __attribute__((ext_vector_type(4))) unsigned int v4u;
typedef __attribute__((ext_vector_type(8))) int v8i;
typedef __attribute__((ext_vector_type(4))) int v4i;

__device__ __forceinline__ void tdm_load_2d(u32 lds_addr, const void* gptr,
                                            u32 tile_d0, u32 tile_d1,
                                            u32 tensor_d0, u32 tensor_d1,
                                            u32 d0_stride, u32 pad_int, u32 pad_amt) {
  u64 ga = (u64)gptr;
  v4u g0 = {1u,                                   // count=1 (valid user D#)
            lds_addr,                             // lds_addr[31:0]
            (u32)ga,                              // global_addr[31:0]
            (u32)((ga >> 32) & 0x1FFFFFFull) | (2u << 30)};  // addr hi | type=2
  u32 w0 = (1u << 16);                            // data_size = 1 -> 2 bytes
  if (pad_amt | pad_int)
    w0 |= (1u << 20) | (pad_int << 22) | (pad_amt << 25);   // pad_enable/interval/amount
  v8i g1;
  g1[0] = (int)w0;                                // workgroup_mask=0 (not in cluster)
  g1[1] = (int)((tensor_d0 & 0xFFFFu) << 16);     // tensor_dim0 lo16 @ bit48
  g1[2] = (int)(((tensor_d0 >> 16) & 0xFFFFu) | ((tensor_d1 & 0xFFFFu) << 16));
  g1[3] = (int)(((tensor_d1 >> 16) & 0xFFFFu) | ((tile_d0 & 0xFFFFu) << 16));
  g1[4] = (int)(tile_d1 & 0xFFFFu);               // tile_dim1 (tile_dim2 = 0)
  g1[5] = (int)d0_stride;                         // tensor_dim0_stride[31:0]
  g1[6] = 0;
  g1[7] = 0;
  v4i z = {0, 0, 0, 0};
#if defined(__clang_major__) && __clang_major__ >= 23
  v8i z8 = {0, 0, 0, 0, 0, 0, 0, 0};
  __builtin_amdgcn_tensor_load_to_lds(g0, g1, z, z, z8, 0);
#else
  __builtin_amdgcn_tensor_load_to_lds(g0, g1, z, z, 0);
#endif
}
#endif  // HAVE_TDM

// ---------------------------------------------------------------------------
// K0: weight pack+transpose  src [K][N] f32 -> dst [N][K] bf16
// ---------------------------------------------------------------------------
__global__ void k_packT(const float* __restrict__ src, __bf16* __restrict__ dst,
                        int K, int N) {
  __shared__ float tile[32][33];
  const int k0 = blockIdx.y * 32, n0 = blockIdx.x * 32;
  const int tx = threadIdx.x, ty = threadIdx.y;
  for (int i = ty; i < 32; i += 8)
    tile[i][tx] = src[(size_t)(k0 + i) * N + n0 + tx];
  __syncthreads();
  for (int i = ty; i < 32; i += 8)
    dst[(size_t)(n0 + i) * K + k0 + tx] = (__bf16)tile[tx][i];
}

// ---------------------------------------------------------------------------
// K1: x [B,C,4096] f32 -> tb [B,4096,C] bf16
// ---------------------------------------------------------------------------
__global__ void k_transpose_cn(const float* __restrict__ x, __bf16* __restrict__ tb) {
  __shared__ float tile[32][33];
  const int b = blockIdx.z;
  const int c0 = blockIdx.x * 32, n0 = blockIdx.y * 32;
  const int tx = threadIdx.x, ty = threadIdx.y;
  for (int i = ty; i < 32; i += 8)
    tile[i][tx] = x[((size_t)(b * 256 + c0 + i)) * 4096 + n0 + tx];
  __syncthreads();
  for (int i = ty; i < 32; i += 8)
    tb[((size_t)b * 4096 + n0 + i) * 256 + c0 + tx] = (__bf16)tile[tx][i];
}

// ---------------------------------------------------------------------------
// K2: QKV GEMM out[32768,768] = tb @ Wqkv^T(+b); relu*scale(q)/relu(k)
// 64x64 block, 256 thr (8 waves), double-buffered LDS, b128 staging,
// epilogue bounced through LDS for 128-bit global stores.
// ---------------------------------------------------------------------------
#define APAD 40
__global__ void k_gemm_qkv(const __bf16* __restrict__ A, const __bf16* __restrict__ BT,
                           const float* __restrict__ bias, __bf16* __restrict__ out,
                           float scale) {
  __shared__ alignas(16) __bf16 smem[4 * 64 * APAD];
  __bf16* sAb[2] = {smem, smem + 64 * APAD};
  __bf16* sBb[2] = {smem + 2 * 64 * APAD, smem + 3 * 64 * APAD};
  const int t = threadIdx.x, lane = t & 31, w = t >> 5;
  const int m0 = blockIdx.y * 64, n0 = blockIdx.x * 64;
  const int mt = w & 3, nb = (w >> 2) * 2;
  const int lrow = t >> 2, lpart = t & 3;
  const __bf16* gA = A + (size_t)(m0 + lrow) * 256 + lpart * 8;
  const __bf16* gB = BT + (size_t)(n0 + lrow) * 256 + lpart * 8;
  const int ldst = lrow * APAD + lpart * 8;
  v8f acc0 = {}, acc1 = {};
  *(uint4*)(sAb[0] + ldst) = *(const uint4*)gA;
  *(uint4*)(sBb[0] + ldst) = *(const uint4*)gB;
  __syncthreads();
  for (int k0 = 0; k0 < 256; k0 += 32) {
    const int cur = (k0 >> 5) & 1;
    if (k0 + 32 < 256) {
      *(uint4*)(sAb[cur ^ 1] + ldst) = *(const uint4*)(gA + k0 + 32);
      *(uint4*)(sBb[cur ^ 1] + ldst) = *(const uint4*)(gB + k0 + 32);
    }
    v16bf af = frag_a(sAb[cur] + mt * 16 * APAD, APAD, lane);
    v16bf b0 = frag_b(sBb[cur] + nb * 16 * APAD, APAD, lane);
    v16bf b1 = frag_b(sBb[cur] + (nb + 1) * 16 * APAD, APAD, lane);
    acc0 = wmma_bf16(af, b0, acc0);
    acc1 = wmma_bf16(af, b1, acc1);
    __syncthreads();
  }
  // epilogue: stage wave's 16x32 bf16 tile in LDS, store rows as uint4
  __bf16* eb = smem + w * 640;  // 16 rows x 40 (16B-aligned rows)
  const int rr = (lane >> 4) * 8, cc = lane & 15;
#pragma unroll
  for (int j = 0; j < 2; ++j) {
    int col = n0 + (nb + j) * 16 + cc;
    float bia = bias[col];
    v8f& a = (j == 0) ? acc0 : acc1;
#pragma unroll
    for (int r = 0; r < 8; ++r) {
      float v = a[r] + bia;
      if (col < 256)      v = fmaxf(v, 0.f) * scale;
      else if (col < 512) v = fmaxf(v, 0.f);
      eb[(rr + r) * 40 + j * 16 + cc] = (__bf16)v;
    }
  }
  __syncthreads();
  const size_t rbase = (size_t)(m0 + mt * 16);
  const int ncol0 = n0 + nb * 16;
#pragma unroll
  for (int k = 0; k < 2; ++k) {
    int c = lane + k * 32, row = c >> 2, part = c & 3;
    *(uint4*)(out + (rbase + row) * 768 + ncol0 + part * 8) =
        *(uint4*)(eb + row * 40 + part * 8);
  }
}

// ---------------------------------------------------------------------------
// K3: ctx^T[b,h,e,d] = sum_n k[b,h,n,d] v[b,h,n,e]; K=4096 split on 4 waves.
// k/v staged TRANSPOSED ([d][n],[e][n]) via b128 global loads.
// ---------------------------------------------------------------------------
#define CST 136
__global__ void k_ctx(const __bf16* __restrict__ qkv, __bf16* __restrict__ ctxT) {
  __shared__ alignas(16) __bf16 skT[32 * CST];
  __shared__ alignas(16) __bf16 svT[32 * CST];
  __shared__ float red[4 * 1024];
  const int bh = blockIdx.x, b = bh >> 3, h = bh & 7;
  const int t = threadIdx.x, lane = t & 31, w = t >> 5;
  const size_t kOff = (size_t)b * 4096 * 768 + 256 + (size_t)h * 32;
  const size_t vOff = kOff + 256;
  v8f acc[2][2] = {};
  for (int n0 = 0; n0 < 4096; n0 += 128) {
#pragma unroll
    for (int i = 0; i < 4; ++i) {
      int c = t + i * 128, row = c >> 2, part = c & 3;
      uint4 kq = *(const uint4*)(qkv + kOff + (size_t)(n0 + row) * 768 + part * 8);
      uint4 vq = *(const uint4*)(qkv + vOff + (size_t)(n0 + row) * 768 + part * 8);
      const __bf16* kp = (const __bf16*)&kq;
      const __bf16* vp = (const __bf16*)&vq;
#pragma unroll
      for (int j = 0; j < 8; ++j) {
        skT[(part * 8 + j) * CST + row] = kp[j];
        svT[(part * 8 + j) * CST + row] = vp[j];
      }
    }
    __syncthreads();
#pragma unroll
    for (int mt = 0; mt < 2; ++mt) {
      v16bf af = frag_a(skT + mt * 16 * CST + w * 32, CST, lane);
#pragma unroll
      for (int nt = 0; nt < 2; ++nt) {
        v16bf bf = frag_b(svT + nt * 16 * CST + w * 32, CST, lane);
        acc[mt][nt] = wmma_bf16(af, bf, acc[mt][nt]);
      }
    }
    __syncthreads();
  }
  const int rr = (lane >> 4) * 8, cc = lane & 15;
#pragma unroll
  for (int mt = 0; mt < 2; ++mt)
#pragma unroll
    for (int nt = 0; nt < 2; ++nt)
#pragma unroll
      for (int r = 0; r < 8; ++r)
        red[w * 1024 + (mt * 16 + rr + r) * 32 + nt * 16 + cc] = acc[mt][nt][r];
  __syncthreads();
#pragma unroll
  for (int i = 0; i < 8; ++i) {
    int p = t + i * 128;  // p = d*32 + e
    float s = red[p] + red[1024 + p] + red[2048 + p] + red[3072 + p];
    ctxT[(size_t)bh * 1024 + (p & 31) * 32 + (p >> 5)] = (__bf16)s;  // [e][d]
  }
}

// ---------------------------------------------------------------------------
// K4: attn_img[b, h*32+e, n] = sum_d q * ctx; b128 staging + packed stores
// ---------------------------------------------------------------------------
__global__ void k_attn(const __bf16* __restrict__ qkv, const __bf16* __restrict__ ctxT,
                       __bf16* __restrict__ attn) {
  __shared__ alignas(16) __bf16 sq[128 * 48];
  __shared__ alignas(16) __bf16 sc[32 * 40];
  const int bh = blockIdx.x, b = bh >> 3, h = bh & 7;
  const int t = threadIdx.x, lane = t & 31, w = t >> 5;
  const int n0blk = blockIdx.y * 128;
  if (t < 128) {  // sc[e][d], 16B chunks
    int row = t >> 2, part = t & 3;
    *(uint4*)(sc + row * 40 + part * 8) = ((const uint4*)(ctxT + (size_t)bh * 1024))[t];
  }
#pragma unroll
  for (int i = 0; i < 2; ++i) {  // q rows, 16B chunks
    int c = t + i * 256, row = c >> 2, part = c & 3;
    size_t off = ((size_t)(b * 4096 + n0blk + row)) * 768 + h * 32 + part * 8;
    *((uint4*)(sq + row * 48 + part * 8)) = ((const uint4*)qkv)[off >> 3];
  }
  __syncthreads();
  v16bf b0 = frag_b(sc, 40, lane);
  v16bf b1 = frag_b(sc + 16 * 40, 40, lane);
  v16bf af = frag_a(sq + (w * 16) * 48, 48, lane);
  v8f d0 = {}, d1 = {};
  d0 = wmma_bf16(af, b0, d0);
  d1 = wmma_bf16(af, b1, d1);
  const int rr = (lane >> 4) * 8, cc = lane & 15;
  const int nbase = n0blk + w * 16 + rr;
  const size_t obase = (size_t)b * 256 + h * 32;
  v8bf o0, o1;
#pragma unroll
  for (int r = 0; r < 8; ++r) {
    o0[r] = (__bf16)d0[r];
    o1[r] = (__bf16)d1[r];
  }
  *(v8bf*)(attn + (obase + cc) * 4096 + nbase) = o0;
  *(v8bf*)(attn + (obase + 16 + cc) * 4096 + nbase) = o1;
}

// ---------------------------------------------------------------------------
// K5: 3x3 depthwise SAME conv; writes transposed conv_t[(b*4096+n)*256+c]
// ---------------------------------------------------------------------------
__global__ void k_dwconv(const __bf16* __restrict__ img, const float* __restrict__ wgt,
                         const float* __restrict__ bias, __bf16* __restrict__ outT) {
  const int idx = blockIdx.x * 256 + threadIdx.x;
  const int xc = idx & 63, y = (idx >> 6) & 63, c = (idx >> 12) & 255, b = idx >> 20;
  const __bf16* p = img + (size_t)(b * 256 + c) * 4096;
  const float* wk = wgt + c * 9;
  float s = bias[c];
#pragma unroll
  for (int dy = -1; dy <= 1; ++dy) {
    int yy = y + dy;
    if (yy < 0 || yy > 63) continue;
#pragma unroll
    for (int dx = -1; dx <= 1; ++dx) {
      int xx = xc + dx;
      if (xx < 0 || xx > 63) continue;
      s += wk[(dy + 1) * 3 + (dx + 1)] * (float)p[yy * 64 + xx];
    }
  }
  outT[((size_t)b * 4096 + y * 64 + xc) * 256 + c] = (__bf16)s;
}

// ---------------------------------------------------------------------------
// K6: proj GEMM + residual + RepBN1 -> t1 (f32), t1b (bf16)
// double-buffered; vectorized dual-output epilogue via LDS
// ---------------------------------------------------------------------------
__global__ void k_gemm_proj(const __bf16* __restrict__ A, const __bf16* __restrict__ BT,
                            const float* __restrict__ bias, const float* __restrict__ x,
                            const float* __restrict__ g1, const float* __restrict__ be1,
                            const float* __restrict__ mu1, const float* __restrict__ va1,
                            const float* __restrict__ al1,
                            float* __restrict__ t1, __bf16* __restrict__ t1b) {
  __shared__ alignas(16) __bf16 smem[4 * 64 * APAD];
  __bf16* sAb[2] = {smem, smem + 64 * APAD};
  __bf16* sBb[2] = {smem + 2 * 64 * APAD, smem + 3 * 64 * APAD};
  const int t = threadIdx.x, lane = t & 31, w = t >> 5;
  const int m0 = blockIdx.y * 64, n0 = blockIdx.x * 64;
  const int mt = w & 3, nb = (w >> 2) * 2;
  const int lrow = t >> 2, lpart = t & 3;
  const __bf16* gA = A + (size_t)(m0 + lrow) * 256 + lpart * 8;
  const __bf16* gB = BT + (size_t)(n0 + lrow) * 256 + lpart * 8;
  const int ldst = lrow * APAD + lpart * 8;
  v8f acc0 = {}, acc1 = {};
  *(uint4*)(sAb[0] + ldst) = *(const uint4*)gA;
  *(uint4*)(sBb[0] + ldst) = *(const uint4*)gB;
  __syncthreads();
  for (int k0 = 0; k0 < 256; k0 += 32) {
    const int cur = (k0 >> 5) & 1;
    if (k0 + 32 < 256) {
      *(uint4*)(sAb[cur ^ 1] + ldst) = *(const uint4*)(gA + k0 + 32);
      *(uint4*)(sBb[cur ^ 1] + ldst) = *(const uint4*)(gB + k0 + 32);
    }
    v16bf af = frag_a(sAb[cur] + mt * 16 * APAD, APAD, lane);
    v16bf b0 = frag_b(sBb[cur] + nb * 16 * APAD, APAD, lane);
    v16bf b1 = frag_b(sBb[cur] + (nb + 1) * 16 * APAD, APAD, lane);
    acc0 = wmma_bf16(af, b0, acc0);
    acc1 = wmma_bf16(af, b1, acc1);
    __syncthreads();
  }
  // epilogue: residual + RepBN into LDS (f32), then vector stores of t1 & t1b
  float* ep = (float*)smem + w * 576;  // 16 rows x 36 f32 (rows 16B aligned)
  const int rr = (lane >> 4) * 8, cc = lane & 15;
  const float alpha = al1[0];
#pragma unroll
  for (int j = 0; j < 2; ++j) {
    int col = n0 + (nb + j) * 16 + cc;
    float bia = bias[col];
    float ig = rsqrtf(va1[col] + EPS) * g1[col];
    float mu = mu1[col], bb = be1[col];
    v8f& a = (j == 0) ? acc0 : acc1;
#pragma unroll
    for (int r = 0; r < 8; ++r) {
      int row = m0 + mt * 16 + rr + r;
      int bb2 = row >> 12, n = row & 4095;
      float res = x[((size_t)bb2 * 256 + col) * 4096 + n];
      float s = a[r] + bia + res;
      ep[(rr + r) * 36 + j * 16 + cc] = (s - mu) * ig + bb + alpha * s;
    }
  }
  __syncthreads();
  const size_t rbase = (size_t)(m0 + mt * 16);
  const int ncol0 = n0 + nb * 16;
#pragma unroll
  for (int k = 0; k < 4; ++k) {  // t1: 128-bit f32 stores
    int c = lane + k * 32, row = c >> 3, part = c & 7;
    *(v4f*)(t1 + (rbase + row) * 256 + ncol0 + part * 4) =
        *(v4f*)(ep + row * 36 + part * 4);
  }
#pragma unroll
  for (int k = 0; k < 2; ++k) {  // t1b: 128-bit bf16 stores
    int c = lane + k * 32, row = c >> 2, part = c & 3;
    const float* pv = ep + row * 36 + part * 8;
    v8bf o;
#pragma unroll
    for (int e = 0; e < 8; ++e) o[e] = (__bf16)pv[e];
    *(v8bf*)(t1b + (rbase + row) * 256 + ncol0 + part * 8) = o;
  }
}

// ---------------------------------------------------------------------------
// K7: fused FFN (gelu(t1@Wfc1+b1)@Wfc2+b2, residual, RepBN2, [B,C,H,W] store)
// 32-row block, 128 thr (4 waves); weight chunks staged by the TDM
// (tensor_load_to_lds + s_wait_tensorcnt) when available.
// ---------------------------------------------------------------------------
__global__ void k_ffn(const __bf16* __restrict__ t1b, const float* __restrict__ t1,
                      const __bf16* __restrict__ w1T, const float* __restrict__ bfc1,
                      const __bf16* __restrict__ w2T, const float* __restrict__ bfc2,
                      const float* __restrict__ g2, const float* __restrict__ be2,
                      const float* __restrict__ mu2, const float* __restrict__ va2,
                      const float* __restrict__ al2, float* __restrict__ out) {
  __shared__ alignas(16) __bf16 sA[32 * 264];     // t1 rows   [32 m][256 k]
  __shared__ alignas(16) __bf16 sW1T[32 * 264];   // fc1 chunk [32 n][256 k]
  __shared__ alignas(16) __bf16 sH[32 * 40];      // hidden    [32 m][32 k]
  __shared__ alignas(16) __bf16 sW2T[256 * 40];   // fc2 chunk [256 n][32 k]
  const int t = threadIdx.x, lane = t & 31, w = t >> 5;
  const int m0 = blockIdx.x * 32;
#pragma unroll
  for (int i = 0; i < 8; ++i) {  // A tile: 16B chunks
    int c = t + i * 128, row = c >> 5, part = c & 31;
    *(uint4*)(sA + row * 264 + part * 8) =
        *(const uint4*)(t1b + (size_t)(m0 + row) * 256 + part * 8);
  }
  v8f acc[8] = {};
  const int hmt = w & 1, hnt = w >> 1;          // hidden tile per wave
  const int omt = w & 1, onb = (w >> 1) * 8;    // 8 output tiles per wave
  for (int cm0 = 0; cm0 < 2048; cm0 += 32) {
    __syncthreads();
#ifdef HAVE_TDM
    if (t < 32) {  // wave 0 drives the Tensor Data Mover for both weight tiles
      // fc1 chunk: rows cm0..cm0+31 of [2048][256]; LDS pad: every 128 dwords
      // (code 6) add 4 dwords (code 3) -> row stride 264 bf16
      tdm_load_2d((u32)(size_t)(void*)sW1T, w1T + (size_t)cm0 * 256,
                  256, 32, 256, 2048, 256, 6, 3);
      // fc2 chunk: cols cm0..cm0+31 of [256][2048]; pad: every 16 dwords
      // (code 3) add 4 dwords (code 3) -> row stride 40 bf16
      tdm_load_2d((u32)(size_t)(void*)sW2T, w2T + cm0,
                  32, 256, 2048, 256, 2048, 3, 3);
      __builtin_amdgcn_s_wait_tensorcnt(0);
    }
#else
#pragma unroll
    for (int i = 0; i < 8; ++i) {
      int c = t + i * 128, row = c >> 5, part = c & 31;
      *(uint4*)(sW1T + row * 264 + part * 8) =
          *(const uint4*)(w1T + (size_t)(cm0 + row) * 256 + part * 8);
    }
#pragma unroll
    for (int i = 0; i < 8; ++i) {
      int c = t + i * 128, row = c >> 2, part = c & 3;
      *(uint4*)(sW2T + row * 40 + part * 8) =
          *(const uint4*)(w2T + (size_t)row * 2048 + cm0 + part * 8);
    }
#endif
    if (cm0 + 32 < 2048)  // stream next fc1 chunk toward L2 (global_prefetch_b8)
      __builtin_prefetch(w1T + (size_t)(cm0 + 32 + (t >> 2)) * 256 + (t & 3) * 64, 0, 1);
    __syncthreads();
    // GEMM1: h[16x16] = A[16x256] @ W1chunk, K=256
    v8f h = {};
#pragma unroll
    for (int k0 = 0; k0 < 256; k0 += 32) {
      v16bf af = frag_a(sA + hmt * 16 * 264 + k0, 264, lane);
      v16bf bf = frag_b(sW1T + hnt * 16 * 264 + k0, 264, lane);
      h = wmma_bf16(af, bf, h);
    }
    {
      const int rr = (lane >> 4) * 8, cc = lane & 15;
      const int col = hnt * 16 + cc;
      const float bia = bfc1[cm0 + col];
#pragma unroll
      for (int r = 0; r < 8; ++r) {
        float xv = h[r] + bia;
        float ge = 0.5f * xv * (1.0f + erff(xv * 0.70710678118f));  // exact GELU
        sH[(hmt * 16 + rr + r) * 40 + col] = (__bf16)ge;
      }
    }
    __syncthreads();
    // GEMM2: acc += H[16x32] @ W2chunk[32x256]
    v16bf af2 = frag_a(sH + omt * 16 * 40, 40, lane);
#pragma unroll
    for (int j = 0; j < 8; ++j) {
      v16bf bf2 = frag_b(sW2T + (onb + j) * 16 * 40, 40, lane);
      acc[j] = wmma_bf16(af2, bf2, acc[j]);
    }
  }
  const int rr = (lane >> 4) * 8, cc = lane & 15;
  const float alpha = al2[0];
  const int rowb = m0 + omt * 16 + rr;
  const int b = rowb >> 12, nb0 = rowb & 4095;
#pragma unroll
  for (int j = 0; j < 8; ++j) {
    int col = (onb + j) * 16 + cc;
    float bia = bfc2[col];
    float ig = rsqrtf(va2[col] + EPS) * g2[col];
    float mu = mu2[col], bb = be2[col];
    float o[8];
#pragma unroll
    for (int r = 0; r < 8; ++r) {
      float s = t1[(size_t)(rowb + r) * 256 + col] + acc[j][r] + bia;
      o[r] = (s - mu) * ig + bb + alpha * s;
    }
    float* op = out + ((size_t)b * 256 + col) * 4096 + nb0;
    *(v4f*)op = (v4f){o[0], o[1], o[2], o[3]};
    *(v4f*)(op + 4) = (v4f){o[4], o[5], o[6], o[7]};
  }
}

// ---------------------------------------------------------------------------
// Host launcher
// ---------------------------------------------------------------------------
extern "C" void kernel_launch(void* const* d_in, const int* in_sizes, int n_in,
                              void* d_out, int out_size, void* d_ws, size_t ws_size,
                              hipStream_t stream) {
  (void)in_sizes; (void)n_in; (void)out_size; (void)ws_size;
  const float* x     = (const float*)d_in[0];
  const float* Wqkv  = (const float*)d_in[1];
  const float* bqkv  = (const float*)d_in[2];
  const float* dw_w  = (const float*)d_in[3];
  const float* dw_b  = (const float*)d_in[4];
  const float* Wproj = (const float*)d_in[5];
  const float* bproj = (const float*)d_in[6];
  const float* bn1_g = (const float*)d_in[7];
  const float* bn1_b = (const float*)d_in[8];
  const float* bn1_m = (const float*)d_in[9];
  const float* bn1_v = (const float*)d_in[10];
  const float* a1    = (const float*)d_in[11];
  const float* Wfc1  = (const float*)d_in[12];
  const float* bfc1  = (const float*)d_in[13];
  const float* Wfc2  = (const float*)d_in[14];
  const float* bfc2  = (const float*)d_in[15];
  const float* bn2_g = (const float*)d_in[16];
  const float* bn2_b = (const float*)d_in[17];
  const float* bn2_m = (const float*)d_in[18];
  const float* bn2_v = (const float*)d_in[19];
  const float* a2    = (const float*)d_in[20];

  char* ws = (char*)d_ws;
  size_t off = 0;
  auto alloc = [&](size_t bytes) -> void* {
    void* p = ws + off;
    off = (off + bytes + 255) & ~(size_t)255;
    return p;
  };
  const size_t M = 32768;
  __bf16* tb    = (__bf16*)alloc(M * 256 * 2);
  __bf16* wqkvT = (__bf16*)alloc((size_t)768 * 256 * 2);
  __bf16* wprjT = (__bf16*)alloc((size_t)256 * 256 * 2);
  __bf16* wfc1T = (__bf16*)alloc((size_t)2048 * 256 * 2);
  __bf16* wfc2T = (__bf16*)alloc((size_t)256 * 2048 * 2);
  __bf16* qkv   = (__bf16*)alloc(M * 768 * 2);
  __bf16* ctxT  = (__bf16*)alloc(64 * 1024 * 2);
  __bf16* attn  = (__bf16*)alloc((size_t)8 * 256 * 4096 * 2);
  __bf16* convt = (__bf16*)alloc(M * 256 * 2);
  float*  t1    = (float*)alloc(M * 256 * 4);
  __bf16* t1b   = (__bf16*)alloc(M * 256 * 2);

  dim3 tb32(32, 8);
  k_packT<<<dim3(24, 8), tb32, 0, stream>>>(Wqkv,  wqkvT, 256, 768);
  k_packT<<<dim3(8, 8),  tb32, 0, stream>>>(Wproj, wprjT, 256, 256);
  k_packT<<<dim3(64, 8), tb32, 0, stream>>>(Wfc1,  wfc1T, 256, 2048);
  k_packT<<<dim3(8, 64), tb32, 0, stream>>>(Wfc2,  wfc2T, 2048, 256);

  k_transpose_cn<<<dim3(8, 128, 8), tb32, 0, stream>>>(x, tb);

  k_gemm_qkv<<<dim3(12, 512), 256, 0, stream>>>(tb, wqkvT, bqkv, qkv,
                                                0.17677669529663687f /* 32^-0.5 */);
  k_ctx<<<64, 128, 0, stream>>>(qkv, ctxT);
  k_attn<<<dim3(64, 32), 256, 0, stream>>>(qkv, ctxT, attn);
  k_dwconv<<<32768, 256, 0, stream>>>(attn, dw_w, dw_b, convt);
  k_gemm_proj<<<dim3(4, 512), 256, 0, stream>>>(convt, wprjT, bproj, x,
                                                bn1_g, bn1_b, bn1_m, bn1_v, a1,
                                                t1, t1b);
  k_ffn<<<1024, 128, 0, stream>>>(t1b, t1, wfc1T, bfc1, wfc2T, bfc2,
                                  bn2_g, bn2_b, bn2_m, bn2_v, a2, (float*)d_out);
}
